// TacticalGNN_54924041781941
// MI455X (gfx1250) — compile-verified, matrix operand order checked by reading, not run on hardware
//
#include <hip/hip_runtime.h>

// ---------------------------------------------------------------------------
// Types for CDNA5 WMMA (wave32): 16x16x32 bf16 -> f32
// ---------------------------------------------------------------------------
typedef __attribute__((ext_vector_type(16))) __bf16        v16bf;
typedef __attribute__((ext_vector_type(8)))  float         v8f;
typedef __attribute__((ext_vector_type(4)))  unsigned int  v4u;

union Frag16 {
    v16bf v;      // 16 bf16 = 8 VGPRs per lane
    v4u   q[2];   // two 16-byte load chunks
};

__device__ __forceinline__ unsigned short f2bf(float f) {
    unsigned u = __float_as_uint(f);
    unsigned r = u + 0x7fffu + ((u >> 16) & 1u);   // round-to-nearest-even
    return (unsigned short)(r >> 16);
}

// Order-preserving uint encoding of float for atomicMax-based segment max.
__device__ __forceinline__ unsigned enc_f32(float f) {
    unsigned u = __float_as_uint(f);
    return (u & 0x80000000u) ? ~u : (u ^ 0x80000000u);
}
__device__ __forceinline__ float dec_f32(unsigned k) {
    if (k == 0u) return 0.0f;  // untouched segment -> reference's isfinite fallback
    unsigned u = (k & 0x80000000u) ? (k ^ 0x80000000u) : ~k;
    return __uint_as_float(u);
}

// ---------------------------------------------------------------------------
// WMMA GEMM: out[M,128] = A_bf16[M,Kpad] @ W_bf16[Kpad,128] + bias[128]
// A row-major (lda = Kpad), B pre-transposed: Bt[n, k] (ldb = Kpad).
// One block = 8 waves; wave w computes the 16x16 tile at columns [16w,16w+16).
// Fragment layout per ISA 7.12.2 (16-bit A 16x32): lane<16 holds row M=lane,
// K = {k0..k0+7, k0+16..k0+23}; lane>=16 holds K = {k0+8..15, k0+24..31}.
// B uses the mirrored (column) layout via the transposed weight buffer.
// ---------------------------------------------------------------------------
__global__ __launch_bounds__(256) void gemm_wmma_bf16(
    const unsigned short* __restrict__ A, int lda,
    const unsigned short* __restrict__ Bt,
    const float* __restrict__ bias,
    float* __restrict__ out, int M, int ktiles)
{
    const int m0   = blockIdx.x * 16;
    if (m0 >= M) return;
    const int lane = threadIdx.x & 31;
    const int wave = threadIdx.x >> 5;
    const int nb   = wave * 16;          // 8 waves cover all 128 output cols
    const int row  = lane & 15;
    const int half = lane >> 4;

    v8f c;
    const float bv = bias[nb + row];     // N = nb+row for this lane's C/D slots
    #pragma unroll
    for (int r = 0; r < 8; ++r) c[r] = bv;

    const int mr = (m0 + row < M) ? (m0 + row) : (M - 1);   // read clamp
    const unsigned short* ap = A  + (size_t)mr * lda        + half * 8;
    const unsigned short* bp = Bt + (size_t)(nb + row) * lda + half * 8;

    for (int t = 0; t < ktiles; ++t) {
        Frag16 a, b;
        a.q[0] = *(const v4u*)(ap);        // k0 + half*8      .. +7
        a.q[1] = *(const v4u*)(ap + 16);   // k0 + 16 + half*8 .. +7
        b.q[0] = *(const v4u*)(bp);
        b.q[1] = *(const v4u*)(bp + 16);
        c = __builtin_amdgcn_wmma_f32_16x16x32_bf16(
                false, a.v, false, b.v, (short)0, c, false, false);
        ap += 32; bp += 32;
    }

    const int ncol  = nb + row;
    const int mbase = m0 + half * 8;      // VGPR r -> M = r + half*8
    #pragma unroll
    for (int r = 0; r < 8; ++r)
        if (mbase + r < M)
            out[(size_t)(mbase + r) * 128 + ncol] = c[r];
}

// ---------------------------------------------------------------------------
// Weight prep: Bt[n*Kpad + k] = bf16(W[k*128 + n]), zero-padded to Kpad.
// ---------------------------------------------------------------------------
__global__ void wconv_k(const float* __restrict__ W, int K, int Kpad,
                        unsigned short* __restrict__ Bt)
{
    int i = blockIdx.x * blockDim.x + threadIdx.x;
    if (i >= 128 * Kpad) return;
    int n = i / Kpad, k = i - n * Kpad;
    Bt[i] = (k < K) ? f2bf(W[(size_t)k * 128 + n]) : (unsigned short)0;
}

// x[N,7] f32 -> xb[N,32] bf16 (K padded to one WMMA step)
__global__ void xconv_k(const float* __restrict__ x, int N,
                        unsigned short* __restrict__ xb)
{
    int i = blockIdx.x * blockDim.x + threadIdx.x;
    if (i >= N * 32) return;
    int n = i >> 5, k = i & 31;
    xb[i] = (k < 7) ? f2bf(x[n * 7 + k]) : (unsigned short)0;
}

// ---------------------------------------------------------------------------
// Edge phase. One wave (32 lanes) per edge; H*C == 128 in every layer.
// ---------------------------------------------------------------------------
__global__ __launch_bounds__(256) void edge_logits_k(
    const float* __restrict__ xl, const float* __restrict__ xr,
    const float* __restrict__ att,
    const int* __restrict__ esrc, const int* __restrict__ edst,
    int E, int Etot, int H, int C,
    float* __restrict__ logits, unsigned* __restrict__ amax)
{
    const int wave = threadIdx.x >> 5;
    const int lane = threadIdx.x & 31;
    const int e = blockIdx.x * 8 + wave;
    if (e >= Etot) return;
    const int s = (e < E) ? esrc[e] : (e - E);   // appended self-loops
    const int d = (e < E) ? edst[e] : (e - E);
    const int HC = H * C;
    const float* pl = xl + (size_t)s * HC;
    const float* pr = xr + (size_t)d * HC;
    for (int h = 0; h < H; ++h) {
        float acc = 0.0f;
        for (int cc = lane; cc < C; cc += 32) {
            float v = pl[h * C + cc] + pr[h * C + cc];
            v = (v > 0.0f) ? v : 0.2f * v;       // leaky_relu(0.2)
            acc += v * att[h * C + cc];
        }
        #pragma unroll
        for (int m = 16; m > 0; m >>= 1) acc += __shfl_xor(acc, m, 32);
        if (lane == 0) {
            logits[(size_t)e * H + h] = acc;
            atomicMax(&amax[(size_t)d * H + h], enc_f32(acc));
        }
    }
}

__global__ void edge_exp_k(float* __restrict__ logits /*in: logit, out: exp*/,
                           const unsigned* __restrict__ amax,
                           const int* __restrict__ edst,
                           int E, int Etot, int H,
                           float* __restrict__ denom)
{
    int i = blockIdx.x * blockDim.x + threadIdx.x;
    if (i >= Etot * H) return;
    int e = i / H, h = i - e * H;
    int d = (e < E) ? edst[e] : (e - E);
    float mx = dec_f32(amax[(size_t)d * H + h]);
    float ex = __expf(logits[i] - mx);
    logits[i] = ex;
    atomicAdd(&denom[(size_t)d * H + h], ex);
}

__global__ __launch_bounds__(256) void edge_scatter_k(
    const float* __restrict__ ex, const float* __restrict__ denom,
    const float* __restrict__ xl,
    const int* __restrict__ esrc, const int* __restrict__ edst,
    int E, int Etot, int H, int C, float* __restrict__ acc)
{
    const int wave = threadIdx.x >> 5;
    const int lane = threadIdx.x & 31;
    const int e = blockIdx.x * 8 + wave;
    if (e >= Etot) return;
    const int s = (e < E) ? esrc[e] : (e - E);
    const int d = (e < E) ? edst[e] : (e - E);
    const int HC = H * C;
    const float* pl = xl + (size_t)s * HC;
    float* pa = acc + (size_t)d * HC;
    for (int h = 0; h < H; ++h) {
        float dn = denom[(size_t)d * H + h];
        float alpha = ex[(size_t)e * H + h] / fmaxf(dn, 1e-16f);
        for (int cc = lane; cc < C; cc += 32)
            atomicAdd(&pa[h * C + cc], alpha * pl[h * C + cc]);
    }
}

// mean over heads + bias (+ELU); emits next layer's bf16 input and/or f32 out.
__global__ void node_finalize_k(const float* __restrict__ acc,
                                const float* __restrict__ bias,
                                int N, int H, int C, int do_elu,
                                unsigned short* __restrict__ out_bf,
                                float* __restrict__ out_f32)
{
    int i = blockIdx.x * blockDim.x + threadIdx.x;
    if (i >= N * C) return;
    int n = i / C, cc = i - n * C;
    float v = 0.0f;
    for (int h = 0; h < H; ++h) v += acc[(size_t)n * H * C + h * C + cc];
    v = v / (float)H + bias[cc];
    if (do_elu) v = (v > 0.0f) ? v : (__expf(v) - 1.0f);
    if (out_bf)  out_bf[(size_t)n * C + cc] = f2bf(v);
    if (out_f32) out_f32[(size_t)n * C + cc] = v;
}

// global mean pool (atomic accumulation into [G,128] + counts)
__global__ void pool_k(const float* __restrict__ h3, const int* __restrict__ batch,
                       int N, float* __restrict__ sums, float* __restrict__ cnt)
{
    int i = blockIdx.x * blockDim.x + threadIdx.x;
    if (i >= N * 128) return;
    int n = i >> 7, cc = i & 127;
    int g = batch[n];
    atomicAdd(&sums[(size_t)g * 128 + cc], h3[i]);
    if (cc == 0) atomicAdd(&cnt[g], 1.0f);
}

__global__ void classifier_k(const float* __restrict__ sums,
                             const float* __restrict__ cnt,
                             const float* __restrict__ Wc,
                             const float* __restrict__ bc,
                             int G, float* __restrict__ out)
{
    int i = blockIdx.x * blockDim.x + threadIdx.x;
    if (i >= G * 11) return;
    int g = i / 11, j = i - g * 11;
    float inv = 1.0f / fmaxf(cnt[g], 1.0f);
    float a = bc[j];
    const float* sg = sums + (size_t)g * 128;
    for (int cc = 0; cc < 128; ++cc) a += sg[cc] * inv * Wc[cc * 11 + j];
    out[i] = a;
}

// ---------------------------------------------------------------------------
// Host driver (graph-capture safe: only kernel launches + hipMemsetAsync).
// ---------------------------------------------------------------------------
extern "C" void kernel_launch(void* const* d_in, const int* in_sizes, int n_in,
                              void* d_out, int out_size, void* d_ws, size_t ws_size,
                              hipStream_t stream)
{
    const float* x    = (const float*)d_in[0];
    const int*   eidx = (const int*)  d_in[1];
    const int*   batch= (const int*)  d_in[2];
    const float* W1l  = (const float*)d_in[3];  const float* b1l  = (const float*)d_in[4];
    const float* W1r  = (const float*)d_in[5];  const float* b1r  = (const float*)d_in[6];
    const float* att1 = (const float*)d_in[7];  const float* bias1= (const float*)d_in[8];
    const float* W2l  = (const float*)d_in[9];  const float* b2l  = (const float*)d_in[10];
    const float* W2r  = (const float*)d_in[11]; const float* b2r  = (const float*)d_in[12];
    const float* att2 = (const float*)d_in[13]; const float* bias2= (const float*)d_in[14];
    const float* W3l  = (const float*)d_in[15]; const float* b3l  = (const float*)d_in[16];
    const float* W3r  = (const float*)d_in[17]; const float* b3r  = (const float*)d_in[18];
    const float* att3 = (const float*)d_in[19]; const float* bias3= (const float*)d_in[20];
    const float* Wc   = (const float*)d_in[21]; const float* bc   = (const float*)d_in[22];

    const int N    = in_sizes[0] / 7;
    const int E    = in_sizes[1] / 2;
    const int Etot = E + N;               // + self-loops
    const int G    = out_size / 11;

    const int* esrc = eidx;
    const int* edst = eidx + E;

    size_t off = 0;
    auto alloc = [&](size_t bytes) -> void* {
        off = (off + 255) & ~(size_t)255;
        void* p = (char*)d_ws + off;
        off += bytes;
        return p;
    };

    unsigned short* xb   = (unsigned short*)alloc((size_t)N * 32 * 2);
    unsigned short* hb   = (unsigned short*)alloc((size_t)N * 64 * 2);
    unsigned short* Bt1l = (unsigned short*)alloc(128 * 32 * 2);
    unsigned short* Bt1r = (unsigned short*)alloc(128 * 32 * 2);
    unsigned short* Bt2l = (unsigned short*)alloc(128 * 64 * 2);
    unsigned short* Bt2r = (unsigned short*)alloc(128 * 64 * 2);
    unsigned short* Bt3l = (unsigned short*)alloc(128 * 64 * 2);
    unsigned short* Bt3r = (unsigned short*)alloc(128 * 64 * 2);
    float*    xl    = (float*)   alloc((size_t)N * 128 * 4);
    float*    xr    = (float*)   alloc((size_t)N * 128 * 4);
    float*    accb  = (float*)   alloc((size_t)N * 128 * 4);
    float*    h3    = (float*)   alloc((size_t)N * 128 * 4);
    float*    exb   = (float*)   alloc((size_t)Etot * 2 * 4);
    unsigned* amax  = (unsigned*)alloc((size_t)N * 2 * 4);
    float*    denom = (float*)   alloc((size_t)N * 2 * 4);
    float*    sums  = (float*)   alloc((size_t)G * 128 * 4);
    float*    cnt   = (float*)   alloc((size_t)G * 4);

    // ---- weight / input prep (tiny) ----
    wconv_k<<<(128 * 32 + 255) / 256, 256, 0, stream>>>(W1l, 7,  32, Bt1l);
    wconv_k<<<(128 * 32 + 255) / 256, 256, 0, stream>>>(W1r, 7,  32, Bt1r);
    wconv_k<<<(128 * 64 + 255) / 256, 256, 0, stream>>>(W2l, 64, 64, Bt2l);
    wconv_k<<<(128 * 64 + 255) / 256, 256, 0, stream>>>(W2r, 64, 64, Bt2r);
    wconv_k<<<(128 * 64 + 255) / 256, 256, 0, stream>>>(W3l, 64, 64, Bt3l);
    wconv_k<<<(128 * 64 + 255) / 256, 256, 0, stream>>>(W3r, 64, 64, Bt3r);
    xconv_k<<<(N * 32 + 255) / 256, 256, 0, stream>>>(x, N, xb);

    hipMemsetAsync(sums, 0, (size_t)G * 128 * 4, stream);
    hipMemsetAsync(cnt,  0, (size_t)G * 4, stream);

    const int mg = (N + 15) / 16;          // one 16-row tile per block
    const int eg = (Etot + 7) / 8;         // 8 edge-waves per block

    auto run_layer = [&](const unsigned short* Ain, int Kpad, int ktiles,
                         const unsigned short* Btl, const float* bl,
                         const unsigned short* Btr, const float* br,
                         const float* att, const float* bias,
                         int H, int C, int do_elu,
                         unsigned short* out_bf, float* out_f32) {
        hipMemsetAsync(amax,  0, (size_t)N * 2 * 4, stream);
        hipMemsetAsync(denom, 0, (size_t)N * 2 * 4, stream);
        hipMemsetAsync(accb,  0, (size_t)N * 128 * 4, stream);
        gemm_wmma_bf16<<<mg, 256, 0, stream>>>(Ain, Kpad, Btl, bl, xl, N, ktiles);
        gemm_wmma_bf16<<<mg, 256, 0, stream>>>(Ain, Kpad, Btr, br, xr, N, ktiles);
        edge_logits_k<<<eg, 256, 0, stream>>>(xl, xr, att, esrc, edst,
                                              E, Etot, H, C, exb, amax);
        edge_exp_k<<<(Etot * H + 255) / 256, 256, 0, stream>>>(exb, amax, edst,
                                                               E, Etot, H, denom);
        edge_scatter_k<<<eg, 256, 0, stream>>>(exb, denom, xl, esrc, edst,
                                               E, Etot, H, C, accb);
        node_finalize_k<<<(N * C + 255) / 256, 256, 0, stream>>>(
            accb, bias, N, H, C, do_elu, out_bf, out_f32);
    };

    // layer 1: 7 -> (H=2, C=64), ELU
    run_layer(xb, 32, 1, Bt1l, b1l, Bt1r, b1r, att1, bias1, 2, 64, 1, hb, nullptr);
    // layer 2: 64 -> (H=2, C=64), ELU   (reads hb, then overwrites it)
    run_layer(hb, 64, 2, Bt2l, b2l, Bt2r, b2r, att2, bias2, 2, 64, 1, hb, nullptr);
    // layer 3: 64 -> (H=1, C=128), no ELU
    run_layer(hb, 64, 2, Bt3l, b3l, Bt3r, b3r, att3, bias3, 1, 128, 0, nullptr, h3);

    // global mean pool + classifier
    pool_k<<<(N * 128 + 255) / 256, 256, 0, stream>>>(h3, batch, N, sums, cnt);
    classifier_k<<<(G * 11 + 255) / 256, 256, 0, stream>>>(sums, cnt, Wc, bc, G,
                                                           (float*)d_out);
}
